// TransformerEncoder_7164005450383
// MI455X (gfx1250) — compile-verified, matrix operand order checked by reading, not run on hardware
//
#include <hip/hip_runtime.h>
#include <hip/hip_bf16.h>
#include <math.h>

// ---------------- problem constants ----------------
constexpr int NLc = 2, Bc = 8, Lc = 128, Hc = 256, NHc = 4, HDc = 64, INNERc = 1024;
constexpr float SCALEc = 0.125f;            // 1/sqrt(64)
constexpr float MASKVAL = -1e9f;

// ---------------- WMMA types ----------------
typedef __attribute__((ext_vector_type(16))) __bf16 bf16x16;
typedef __attribute__((ext_vector_type(8)))  float  f32x8;

__device__ __forceinline__ f32x8 wmma_bf16(bf16x16 a, bf16x16 b, f32x8 c) {
    return __builtin_amdgcn_wmma_f32_16x16x32_bf16(false, a, false, b, (short)0, c,
                                                   false, false);
}

// A fragment (16x32 bf16) from row-major f32, per the ISA layout:
// lanes 0-15: row M=lane, K = 0..7 and 16..23 ; lanes 16-31: row M=lane-16, K = 8..15 and 24..31
__device__ __forceinline__ bf16x16 load_a_frag(const float* __restrict__ A, int lda) {
    int lane = threadIdx.x & 31;
    int row  = lane & 15;
    int kb   = (lane >> 4) * 8;
    const float* p = A + row * lda + kb;
    bf16x16 f;
#pragma unroll
    for (int e = 0; e < 8; ++e) f[e]     = (__bf16)p[e];
#pragma unroll
    for (int e = 0; e < 8; ++e) f[8 + e] = (__bf16)p[16 + e];
    return f;
}

// B fragment (32x16 bf16) from row-major f32 (K x N), symmetric layout:
// lane holds column N=lane&15, K groups as in A.
__device__ __forceinline__ bf16x16 load_b_frag(const float* __restrict__ Bm, int ldb) {
    int lane = threadIdx.x & 31;
    int col  = lane & 15;
    int kb   = (lane >> 4) * 8;
    bf16x16 f;
#pragma unroll
    for (int e = 0; e < 8; ++e) f[e]     = (__bf16)Bm[(kb + e) * ldb + col];
#pragma unroll
    for (int e = 0; e < 8; ++e) f[8 + e] = (__bf16)Bm[(kb + 16 + e) * ldb + col];
    return f;
}

// ---------------- generic GEMM: C = A(MxK) @ W(KxN) (+bias) (+gelu) ----------------
// one 16x16 tile per wave; 8 waves / block.
template <int EPILOGUE>   // 0 = none, 1 = exact GELU
__global__ void gemm_wmma_kernel(const float* __restrict__ A, const float* __restrict__ W,
                                 const float* __restrict__ bias, float* __restrict__ C,
                                 int M, int N, int K) {
    int wave  = blockIdx.x * (blockDim.x >> 5) + (threadIdx.x >> 5);
    int tn_ct = N >> 4;
    int tm = wave / tn_ct;
    int tn = wave % tn_ct;
    if (tm >= (M >> 4)) return;

    const float* Ab = A + (size_t)tm * 16 * K;
    f32x8 acc = {};
    for (int k = 0; k < K; k += 32) {
        __builtin_prefetch(Ab + k + 128, 0, 1);   // global_prefetch_b8
        bf16x16 a = load_a_frag(Ab + k, K);
        bf16x16 b = load_b_frag(W + (size_t)k * N + tn * 16, N);
        acc = wmma_bf16(a, b, acc);
    }

    int lane = threadIdx.x & 31;
    int col  = tn * 16 + (lane & 15);
    float bv = bias ? bias[col] : 0.0f;
#pragma unroll
    for (int r = 0; r < 8; ++r) {
        int row = tm * 16 + r + ((lane >> 4) * 8);
        float v = acc[r] + bv;
        if (EPILOGUE == 1) v = 0.5f * v * (1.0f + erff(v * 0.70710678f));
        C[(size_t)row * N + col] = v;
    }
}

// ---------------- fused attention ----------------
// grid: B*NH*(L/16) blocks; block = 256 threads (8 waves).
// q,k,v,ctx layout: (B, L, H) with head h occupying columns [h*HD, h*HD+HD)
__global__ void attention_kernel(const float* __restrict__ q, const float* __restrict__ k,
                                 const float* __restrict__ v, const float* __restrict__ tk,
                                 const float* __restrict__ tv,
                                 const unsigned char* __restrict__ mask,
                                 float* __restrict__ ctx) {
    __shared__ float lds_q[16 * HDc];       // 4 KB
    __shared__ float lds_s[16 * Lc];        // 8 KB  (scores -> probs)
    __shared__ float lds_bias[16 * HDc];    // 4 KB  (probs . tv)

    const int it   = blockIdx.x & 7;
    const int h    = (blockIdx.x >> 3) & 3;
    const int b    = blockIdx.x >> 5;
    const int tid  = threadIdx.x;
    const int wave = tid >> 5;
    const int lane = tid & 31;

    // ---- stage q tile (16 x 64 f32) into LDS ----
    {
        const size_t qbase = ((size_t)(b * Lc + it * 16)) * Hc + h * HDc;
        int idx = tid * 4;                    // 256 threads * float4 = 1024 floats
        int i = idx >> 6, d = idx & 63;
        float4 val = *reinterpret_cast<const float4*>(q + qbase + (size_t)i * Hc + d);
        *reinterpret_cast<float4*>(&lds_q[idx]) = val;
    }
    __syncthreads();

    // ---- phase 1: scores = q@k^T + q.tk, scaled + masked, into LDS ----
    {
        const int jt = wave;                  // 8 waves x 16 cols = 128
        f32x8 acc = {};
        const float* kcol = k + ((size_t)(b * Lc + jt * 16 + (lane & 15))) * Hc + h * HDc;
        for (int ks = 0; ks < HDc; ks += 32) {
            bf16x16 a = load_a_frag(lds_q + ks, HDc);
            bf16x16 bb;
            int kb0 = ks + ((lane >> 4) * 8);
#pragma unroll
            for (int e = 0; e < 8; ++e) bb[e]     = (__bf16)kcol[kb0 + e];
#pragma unroll
            for (int e = 0; e < 8; ++e) bb[8 + e] = (__bf16)kcol[kb0 + 16 + e];
            acc = wmma_bf16(a, bb, acc);
        }
        const int j = jt * 16 + (lane & 15);
#pragma unroll
        for (int r = 0; r < 8; ++r) {
            int il = r + ((lane >> 4) * 8);
            int ig = it * 16 + il;
            const float4* tkp =
                reinterpret_cast<const float4*>(tk + (((size_t)(b * Lc + ig)) * Lc + j) * Hc + h * HDc);
            const float4* qp = reinterpret_cast<const float4*>(&lds_q[il * HDc]);
            float dot = 0.0f;
#pragma unroll
            for (int t = 0; t < 16; ++t) {
                float4 a4 = tkp[t];
                float4 q4 = qp[t];
                dot += a4.x * q4.x + a4.y * q4.y + a4.z * q4.z + a4.w * q4.w;
            }
            float s = (acc[r] + dot) * SCALEc;
            if (mask[((size_t)(b * Lc + ig)) * Lc + j]) s = MASKVAL;
            lds_s[il * Lc + j] = s;
        }
    }
    __syncthreads();

    // ---- phase 2: softmax over j (rows of lds_s); 2 rows per wave, 16 lanes/row ----
    {
        const int row = wave * 2 + (lane >> 4);
        const int l16 = lane & 15;
        float vals[8];
        float m = -3.0e38f;
#pragma unroll
        for (int t = 0; t < 8; ++t) {
            vals[t] = lds_s[row * Lc + l16 + t * 16];
            m = fmaxf(m, vals[t]);
        }
        for (int off = 8; off; off >>= 1) m = fmaxf(m, __shfl_xor(m, off, 16));
        float sum = 0.0f;
#pragma unroll
        for (int t = 0; t < 8; ++t) {
            vals[t] = __expf(vals[t] - m);
            sum += vals[t];
        }
        for (int off = 8; off; off >>= 1) sum += __shfl_xor(sum, off, 16);
        float inv = 1.0f / sum;
#pragma unroll
        for (int t = 0; t < 8; ++t) lds_s[row * Lc + l16 + t * 16] = vals[t] * inv;
    }
    __syncthreads();

    // ---- phase 3: waves 0-3 -> ctx = P@V (WMMA); waves 4-7 -> P.tv bias (HBM-bound VALU) ----
    f32x8 cacc = {};
    if (wave < 4) {
        const int nt  = wave;                // d-tile: 4 x 16 = 64
        const int col = nt * 16 + (lane & 15);
        for (int ks = 0; ks < Lc; ks += 32) {
            bf16x16 a = load_a_frag(lds_s + ks, Lc);
            bf16x16 bb;
            int kb0 = ks + ((lane >> 4) * 8);
#pragma unroll
            for (int e = 0; e < 8; ++e)
                bb[e] = (__bf16)v[((size_t)(b * Lc + kb0 + e)) * Hc + h * HDc + col];
#pragma unroll
            for (int e = 0; e < 8; ++e)
                bb[8 + e] = (__bf16)v[((size_t)(b * Lc + kb0 + 16 + e)) * Hc + h * HDc + col];
            cacc = wmma_bf16(a, bb, cacc);
        }
    } else {
        const int t2 = tid - 128;            // 0..127
        const int i  = t2 >> 3;              // 16 rows
        const int d0 = (t2 & 7) * 8;         // 8 d-groups of 8
        const int ig = it * 16 + i;
        float acc8[8];
#pragma unroll
        for (int t = 0; t < 8; ++t) acc8[t] = 0.0f;
        const float* tvb = tv + (((size_t)(b * Lc + ig)) * Lc) * Hc + h * HDc + d0;
        for (int j = 0; j < Lc; ++j) {
            float p = lds_s[i * Lc + j];
            const float4* pp = reinterpret_cast<const float4*>(tvb + (size_t)j * Hc);
            float4 x0 = pp[0], x1 = pp[1];
            acc8[0] += p * x0.x; acc8[1] += p * x0.y; acc8[2] += p * x0.z; acc8[3] += p * x0.w;
            acc8[4] += p * x1.x; acc8[5] += p * x1.y; acc8[6] += p * x1.z; acc8[7] += p * x1.w;
        }
#pragma unroll
        for (int t = 0; t < 8; ++t) lds_bias[i * HDc + d0 + t] = acc8[t];
    }
    __syncthreads();

    if (wave < 4) {
        const int nt  = wave;
        const int col = nt * 16 + (lane & 15);
#pragma unroll
        for (int r = 0; r < 8; ++r) {
            int il = r + ((lane >> 4) * 8);
            int ig = it * 16 + il;
            ctx[((size_t)(b * Lc + ig)) * Hc + h * HDc + col] = cacc[r] + lds_bias[il * HDc + col];
        }
    }
}

// ---------------- fused bias + residual + LayerNorm (one row / block, H=256) ----------------
__global__ void residual_ln_kernel(const float* __restrict__ y, const float* __restrict__ bias,
                                   const float* __restrict__ res, const float* __restrict__ gamma,
                                   const float* __restrict__ beta, float eps,
                                   float* __restrict__ out) {
    __shared__ float red[256];
    const int row = blockIdx.x;
    const int t   = threadIdx.x;
    const size_t idx = (size_t)row * Hc + t;

    float v = y[idx] + bias[t] + res[idx];
    red[t] = v;
    __syncthreads();
    for (int s = 128; s; s >>= 1) {
        if (t < s) red[t] += red[t + s];
        __syncthreads();
    }
    float mu = red[0] * (1.0f / Hc);
    __syncthreads();

    float d = v - mu;
    red[t] = d * d;
    __syncthreads();
    for (int s = 128; s; s >>= 1) {
        if (t < s) red[t] += red[t + s];
        __syncthreads();
    }
    float var = red[0] * (1.0f / Hc);

    out[idx] = d * rsqrtf(var + eps) * gamma[t] + beta[t];
}

// ---------------- launcher ----------------
extern "C" void kernel_launch(void* const* d_in, const int* in_sizes, int n_in,
                              void* d_out, int out_size, void* d_ws, size_t ws_size,
                              hipStream_t stream) {
    const float* x0  = (const float*)d_in[0];
    const unsigned char* mask = (const unsigned char*)d_in[1];
    const float* tk  = (const float*)d_in[2];
    const float* tv  = (const float*)d_in[3];
    const float* Wq  = (const float*)d_in[4];
    const float* bq  = (const float*)d_in[5];
    const float* Wk  = (const float*)d_in[6];
    const float* bk  = (const float*)d_in[7];
    const float* Wv  = (const float*)d_in[8];
    const float* bv  = (const float*)d_in[9];
    const float* Wo  = (const float*)d_in[10];
    const float* bo  = (const float*)d_in[11];
    const float* ln1s = (const float*)d_in[12];
    const float* ln1b = (const float*)d_in[13];
    const float* W1  = (const float*)d_in[14];
    const float* b1  = (const float*)d_in[15];
    const float* W2  = (const float*)d_in[16];
    const float* b2  = (const float*)d_in[17];
    const float* ln2s = (const float*)d_in[18];
    const float* ln2b = (const float*)d_in[19];

    float* out = (float*)d_out;
    float* ws  = (float*)d_ws;

    const size_t MLH = (size_t)Bc * Lc * Hc;      // 262144
    const int    M   = Bc * Lc;                   // 1024

    float* qb   = ws;
    float* kb_  = qb + MLH;
    float* vb   = kb_ + MLH;
    float* ctx  = vb + MLH;
    float* tmp  = ctx + MLH;
    float* attn = tmp + MLH;
    float* ffn  = attn + MLH;                     // M * INNER floats

    dim3 blk(256);
    auto gemm_blocks = [](int m, int n) { return (unsigned)((m >> 4) * (n >> 4) / 8); };

    for (int l = 0; l < NLc; ++l) {
        const float* x = (l == 0) ? x0 : out + (size_t)(l - 1) * MLH;
        const float* wq = Wq + (size_t)l * Hc * Hc;
        const float* wk = Wk + (size_t)l * Hc * Hc;
        const float* wv = Wv + (size_t)l * Hc * Hc;
        const float* wo = Wo + (size_t)l * Hc * Hc;
        const float* w1 = W1 + (size_t)l * Hc * INNERc;
        const float* w2 = W2 + (size_t)l * INNERc * Hc;

        gemm_wmma_kernel<0><<<gemm_blocks(M, Hc), blk, 0, stream>>>(x, wq, bq + l * Hc, qb, M, Hc, Hc);
        gemm_wmma_kernel<0><<<gemm_blocks(M, Hc), blk, 0, stream>>>(x, wk, bk + l * Hc, kb_, M, Hc, Hc);
        gemm_wmma_kernel<0><<<gemm_blocks(M, Hc), blk, 0, stream>>>(x, wv, bv + l * Hc, vb, M, Hc, Hc);

        attention_kernel<<<dim3(Bc * NHc * (Lc / 16)), blk, 0, stream>>>(qb, kb_, vb, tk, tv, mask, ctx);

        gemm_wmma_kernel<0><<<gemm_blocks(M, Hc), blk, 0, stream>>>(ctx, wo, nullptr, tmp, M, Hc, Hc);
        residual_ln_kernel<<<dim3(M), blk, 0, stream>>>(tmp, bo + l * Hc, x,
                                                        ln1s + l * Hc, ln1b + l * Hc, 1e-5f, attn);

        gemm_wmma_kernel<1><<<gemm_blocks(M, INNERc), blk, 0, stream>>>(attn, w1, b1 + l * INNERc,
                                                                        ffn, M, INNERc, Hc);
        gemm_wmma_kernel<0><<<gemm_blocks(M, Hc), blk, 0, stream>>>(ffn, w2, nullptr, tmp, M, Hc, INNERc);
        residual_ln_kernel<<<dim3(M), blk, 0, stream>>>(tmp, b2 + l * Hc, attn,
                                                        ln2s + l * Hc, ln2b + l * Hc, 1e-12f,
                                                        out + (size_t)l * MLH);
    }
}
